// UnrolledAutoencodingMamba_1640677507189
// MI455X (gfx1250) — matrix operand order repaired and checked by reading, not compiled, not run on hardware
//
#include <hip/hip_runtime.h>
#include <math.h>
#include <stdint.h>

// ---------------- constants from the reference ----------------
constexpr int Vv    = 8000;
constexpr int Dd    = 512;
constexpr int Ll    = 16;
constexpr int Hh    = 16;
constexpr int Pp    = 64;
constexpr int Ss    = 64;
constexpr int Tt    = 512;
constexpr int KC    = 4;      // conv kernel
constexpr int Bb    = 16;
constexpr int INTER = Hh * Pp;              // 1024
constexpr int CONVC = INTER + 2 * Ss;       // 1152 (G=1)
constexpr int PROJ  = 2 * INTER + 2 * Ss + Hh; // 2192
constexpr int NROWS = Bb * Tt;              // 8192
constexpr int NPAD_IN = 2240;               // PROJ padded up to mult of 64
constexpr int D2    = Dd / 2;               // 256
constexpr float EPSV = 1e-5f;

// ---------------- WMMA types ----------------
typedef __attribute__((ext_vector_type(16))) __bf16        v16bf;
typedef __attribute__((ext_vector_type(8)))  float         v8f;
typedef __attribute__((ext_vector_type(4)))  unsigned int  u32x4;

union BFrag { v16bf v; u32x4 q[2]; };

// Optional CDNA5 async global->LDS path (guarded; falls back to plain stores)
#if __has_builtin(__builtin_amdgcn_global_load_async_to_lds_b32) && \
    __has_builtin(__builtin_amdgcn_s_wait_asynccnt)
#define USE_ASYNC_LDS 1
typedef __attribute__((address_space(1))) int gint_t;   // global int*
typedef __attribute__((address_space(3))) int lint_t;   // LDS int*
#else
#define USE_ASYNC_LDS 0
#endif

__device__ __forceinline__ unsigned short f2bf(float f) {
  unsigned int u = __float_as_uint(f);
  u = (u + 0x7FFFu + ((u >> 16) & 1u)) >> 16;   // RNE
  return (unsigned short)u;
}

__device__ __forceinline__ float block_reduce_sum(float v, float* red) {
  #pragma unroll
  for (int off = 16; off > 0; off >>= 1) v += __shfl_down(v, off, 32);
  int w = threadIdx.x >> 5;
  if ((threadIdx.x & 31) == 0) red[w] = v;
  __syncthreads();
  float t = (threadIdx.x < (blockDim.x >> 5)) ? red[threadIdx.x] : 0.f;
  #pragma unroll
  for (int off = 4; off > 0; off >>= 1) t += __shfl_down(t, off, 32);
  if (threadIdx.x == 0) red[0] = t;
  __syncthreads();
  float r = red[0];
  __syncthreads();
  return r;
}

__device__ __forceinline__ float block_reduce_max(float v, float* red) {
  #pragma unroll
  for (int off = 16; off > 0; off >>= 1) v = fmaxf(v, __shfl_down(v, off, 32));
  int w = threadIdx.x >> 5;
  if ((threadIdx.x & 31) == 0) red[w] = v;
  __syncthreads();
  float t = (threadIdx.x < (blockDim.x >> 5)) ? red[threadIdx.x] : -3.0e38f;
  #pragma unroll
  for (int off = 4; off > 0; off >>= 1) t = fmaxf(t, __shfl_down(t, off, 32));
  if (threadIdx.x == 0) red[0] = t;
  __syncthreads();
  float r = red[0];
  __syncthreads();
  return r;
}

// ---------------- bf16 weight conversion with row padding ----------------
__global__ __launch_bounds__(256)
void convert_pad_kernel(const float* __restrict__ src, unsigned short* __restrict__ dst,
                        int N, int Kd, long total) {
  long idx = (long)blockIdx.x * 256 + threadIdx.x;
  if (idx >= total) return;
  long n = idx / Kd;
  long k = idx - n * Kd;
  float v = (n < N) ? src[(size_t)n * Kd + k] : 0.f;
  dst[idx] = f2bf(v);
}

// ---------------- bf16 WMMA GEMM: C[M,N] = A[M,K] * W[N,K]^T ----------------
// Wave tile: 32(M) x 64(N).  Per K-step: 2 A frags + 4 B frags -> 8 wmma
// (all fragment loads issued up front so they clause together and the wmma
//  burst runs back-to-back after one wait).
// EPI: 0 = store, 1 = C += (residual in-place), 2 = +bias[n], 3 = transposed
// For EPI==3: m = b*Tdim + t, out[(b*Vdim + n)*Tdim + t]
template<int EPI>
__global__ __launch_bounds__(256)
void gemm_bf16_wmma(const unsigned short* __restrict__ A,
                    const unsigned short* __restrict__ W,
                    float* __restrict__ C,
                    const float* __restrict__ bias,
                    int M, int N, int Kd, int Npad, int ldc,
                    int Vdim, int Tdim) {
  const int wid  = blockIdx.x * 8 + (threadIdx.x >> 5);
  const int lane = threadIdx.x & 31;
  const int tilesN = Npad >> 6;           // 64-wide N groups
  const int tilesM = M >> 5;              // 32-row M groups
  if (wid >= tilesM * tilesN) return;
  const int tn = wid % tilesN;
  const int tm = wid / tilesN;

  const int mA    = tm * 32 + (lane & 15);
  const int khalf = (lane >> 4) << 3;     // 0 or 8  (A layout per ISA 7.12.2)
  const unsigned short* rowA0 = A + (size_t)mA * Kd;
  const unsigned short* rowA1 = rowA0 + (size_t)16 * Kd;

  const int nB = tn * 64 + (lane & 15);
  const int kb = (lane >> 4) << 4;        // 0 or 16 (unused; B frags load with khalf, same k order as A)
  const unsigned short* rowW0 = W + (size_t)nB * Kd;

  v8f acc[2][4] = {};

  for (int k0 = 0; k0 < Kd; k0 += 32) {
    BFrag a0, a1, b[4];
    // issue all 12 x b128 loads before any wmma so they clause together
    a0.q[0] = *(const u32x4*)(rowA0 + k0 + khalf);
    a0.q[1] = *(const u32x4*)(rowA0 + k0 + 16 + khalf);
    a1.q[0] = *(const u32x4*)(rowA1 + k0 + khalf);
    a1.q[1] = *(const u32x4*)(rowA1 + k0 + 16 + khalf);
    #pragma unroll
    for (int j = 0; j < 4; ++j) {
      const unsigned short* rowW = rowW0 + (size_t)j * 16 * Kd;
      b[j].q[0] = *(const u32x4*)(rowW + k0 + khalf);
      b[j].q[1] = *(const u32x4*)(rowW + k0 + 16 + khalf);
    }
    if (k0 + 32 < Kd) {
      __builtin_prefetch((const void*)(rowA0 + k0 + 32), 0, 3);  // near-cache hint
      __builtin_prefetch((const void*)(rowA1 + k0 + 32), 0, 3);
    }
    #pragma unroll
    for (int j = 0; j < 4; ++j) {
      acc[0][j] = __builtin_amdgcn_wmma_f32_16x16x32_bf16(
          false, a0.v, false, b[j].v, (short)0, acc[0][j], false, false);
      acc[1][j] = __builtin_amdgcn_wmma_f32_16x16x32_bf16(
          false, a1.v, false, b[j].v, (short)0, acc[1][j], false, false);
    }
  }

  const int ncol0 = tn * 64 + (lane & 15);
  #pragma unroll
  for (int i = 0; i < 2; ++i) {
    const int mrow = tm * 32 + i * 16 + ((lane >> 4) << 3);  // + r
    #pragma unroll
    for (int j = 0; j < 4; ++j) {
      int n = ncol0 + j * 16;
      if ((tn * 64 + j * 16) >= N) continue;       // uniform guard (N % 16 == 0)
      #pragma unroll
      for (int r = 0; r < 8; ++r) {
        int m = mrow + r;
        float val = acc[i][j][r];
        if (EPI == 0) {
          C[(size_t)m * ldc + n] = val;
        } else if (EPI == 1) {
          C[(size_t)m * ldc + n] += val;
        } else if (EPI == 2) {
          C[(size_t)m * ldc + n] = val + bias[n];
        } else {
          int b = m / Tdim, t = m - b * Tdim;
          C[((size_t)b * Vdim + n) * Tdim + t] = val;
        }
      }
    }
  }
}

// ---------------- RMSNorm: out = x * rsqrt(mean(x^2)+eps) * w ----------------
__global__ __launch_bounds__(256)
void rmsnorm_kernel(const float* __restrict__ x, long ldx, const float* __restrict__ w,
                    unsigned short* __restrict__ outb, float* __restrict__ outf, int Dn) {
  __shared__ float red[8];
  const long row = blockIdx.x;
  const float* xr = x + row * ldx;
  float ss = 0.f;
  for (int d = threadIdx.x; d < Dn; d += blockDim.x) { float v = xr[d]; ss += v * v; }
  float tot = block_reduce_sum(ss, red);
  float rs = rsqrtf(tot / Dn + EPSV);
  for (int d = threadIdx.x; d < Dn; d += blockDim.x) {
    float v = xr[d] * rs * w[d];
    if (outb) outb[row * Dn + d] = f2bf(v);
    if (outf) outf[row * Dn + d] = v;
  }
}

// ---------------- gated RMSNorm: out_bf16 = rms(y * silu(z)) * w ----------------
__global__ __launch_bounds__(256)
void gated_rmsnorm_kernel(const float* __restrict__ y, const float* __restrict__ z, int ldz,
                          const float* __restrict__ w, unsigned short* __restrict__ outb) {
  __shared__ float g[INTER];
  __shared__ float red[8];
  const long row = blockIdx.x;
  const float* yr = y + row * (long)INTER;
  const float* zr = z + row * (long)ldz;
  float ss = 0.f;
  for (int d = threadIdx.x; d < INTER; d += blockDim.x) {
    float zv = zr[d];
    float gv = yr[d] * (zv / (1.f + expf(-zv)));
    g[d] = gv; ss += gv * gv;
  }
  __syncthreads();
  float tot = block_reduce_sum(ss, red);
  float rs = rsqrtf(tot / INTER + EPSV);
  for (int d = threadIdx.x; d < INTER; d += blockDim.x)
    outb[row * (long)INTER + d] = f2bf(g[d] * rs * w[d]);
}

// ---------------- depthwise causal conv (K=4) + bias + silu ----------------
__global__ __launch_bounds__(256)
void conv_silu_kernel(const float* __restrict__ zx /* rows stride PROJ, col offset pre-applied */,
                      const float* __restrict__ wc, const float* __restrict__ bc,
                      float* __restrict__ out, int Bn, int Tn) {
  long idx = (long)blockIdx.x * 256 + threadIdx.x;
  long total = (long)Bn * Tn * CONVC;
  if (idx >= total) return;
  int c = (int)(idx % CONVC);
  long bt = idx / CONVC;
  int t = (int)(bt % Tn);
  long b = bt / Tn;
  float acc = bc[c];
  #pragma unroll
  for (int j = 0; j < KC; ++j) {
    int tt = t - (KC - 1) + j;
    if (tt >= 0) acc += wc[c * KC + j] * zx[((size_t)b * Tn + tt) * PROJ + c];
  }
  out[idx] = acc / (1.f + expf(-acc));   // silu
}

// ---------------- dt = softplus(dt_raw + bias); dA = exp(-exp(A_log)*dt) --------
__global__ __launch_bounds__(256)
void dtda_kernel(const float* __restrict__ dtraw /* stride PROJ */,
                 const float* __restrict__ dt_bias, const float* __restrict__ A_log,
                 float* __restrict__ dtv, float* __restrict__ dAv, long rows) {
  long idx = (long)blockIdx.x * 256 + threadIdx.x;
  if (idx >= rows * Hh) return;
  int  h = (int)(idx % Hh);
  long r = idx / Hh;
  float x = dtraw[(size_t)r * PROJ + h] + dt_bias[h];
  float sp = (x > 20.f) ? x : log1pf(expf(x));
  dtv[idx] = sp;
  dAv[idx] = expf(-expf(A_log[h]) * sp);
}

// ---------------- sequential SSM scan, one block per (b,h) chain ----------------
__global__ __launch_bounds__(64)
void scan_kernel(const float* __restrict__ xbcc, const float* __restrict__ dtv,
                 const float* __restrict__ dAv, const float* __restrict__ Dp,
                 float* __restrict__ y, int Bn, int Tn) {
  const int bh = blockIdx.x;
  const int b  = bh / Hh;
  const int h  = bh - b * Hh;
  const int p  = threadIdx.x;            // 0..63 (state row)
  float hst[Ss];
  #pragma unroll
  for (int s = 0; s < Ss; ++s) hst[s] = 0.f;
  __shared__ float Bs[Ss], Cs[Ss];
  const float Dh = Dp[h];
  for (int t = 0; t < Tn; ++t) {
    const size_t rb = (size_t)b * Tn + t;
    const float* xrow = xbcc + rb * CONVC;
#if USE_ASYNC_LDS
    // CDNA5 async copy global->LDS for the B/C broadcast vectors.
    // (integer round-trip: flat LDS addresses truncate to the LDS offset, ISA 10.2)
    __builtin_amdgcn_global_load_async_to_lds_b32(
        (gint_t*)(uintptr_t)(xrow + INTER + p),
        (lint_t*)(uintptr_t)&Bs[p], 0, 0);
    __builtin_amdgcn_global_load_async_to_lds_b32(
        (gint_t*)(uintptr_t)(xrow + INTER + Ss + p),
        (lint_t*)(uintptr_t)&Cs[p], 0, 0);
    __builtin_amdgcn_s_wait_asynccnt(0);
#else
    Bs[p] = xrow[INTER + p];
    Cs[p] = xrow[INTER + Ss + p];
#endif
    __syncthreads();
    const float dA = dAv[rb * Hh + h];
    const float dt = dtv[rb * Hh + h];
    const float xv = xrow[h * Pp + p];
    const float coef = dt * xv;
    float acc = 0.f;
    #pragma unroll
    for (int s = 0; s < Ss; ++s) {
      hst[s] = dA * hst[s] + coef * Bs[s];
      acc += hst[s] * Cs[s];
    }
    y[rb * INTER + h * Pp + p] = acc + Dh * xv;
    __syncthreads();
  }
}

// ---------------- embedding gather ----------------
__global__ __launch_bounds__(256)
void embed_kernel(const float* __restrict__ wte, const int* __restrict__ ids,
                  float* __restrict__ x, long rows, int Dn) {
  long idx = (long)blockIdx.x * 256 + threadIdx.x;
  if (idx >= rows * Dn) return;
  long r = idx / Dn;
  int  d = (int)(idx - r * Dn);
  x[idx] = wte[(size_t)ids[r] * Dn + d];
}

// ---------------- circular window from encoder last-token embedding ----------------
__global__ __launch_bounds__(256)
void window_kernel(const float* __restrict__ e /* (B,D) */, unsigned short* __restrict__ win) {
  long idx = (long)blockIdx.x * 256 + threadIdx.x;
  long total = (long)Bb * Tt * D2;
  if (idx >= total) return;
  int  k = (int)(idx % D2);
  long bt = idx / D2;
  int  t = (int)(bt % Tt);
  long b = bt / Tt;
  win[idx] = f2bf(e[b * Dd + ((t + k) % Dd)]);
}

// ---------------- loss: mean over (b,t) of (lse - logit[label]) ----------------
__global__ void zero_kernel(float* p) { if (blockIdx.x == 0 && threadIdx.x == 0) p[0] = 0.f; }

__global__ __launch_bounds__(256)
void loss_kernel(const float* __restrict__ logitsT /* (B,V,T) */,
                 const int* __restrict__ labels, float* __restrict__ acc, int Vn, int Tn) {
  __shared__ float red[8];
  const int bt = blockIdx.x;
  const int b = bt / Tn, t = bt - b * Tn;
  const float* base = logitsT + (size_t)b * Vn * Tn + t;
  float mx = -3.0e38f;
  for (int v = threadIdx.x; v < Vn; v += blockDim.x)
    mx = fmaxf(mx, base[(size_t)v * Tn]);
  mx = block_reduce_max(mx, red);
  float se = 0.f;
  for (int v = threadIdx.x; v < Vn; v += blockDim.x)
    se += expf(base[(size_t)v * Tn] - mx);
  se = block_reduce_sum(se, red);
  if (threadIdx.x == 0) {
    float lse = mx + logf(se);
    float lv  = base[(size_t)labels[bt] * Tn];
    atomicAdd(acc, lse - lv);
  }
}

__global__ void finalize_loss(const float* acc, float* out, float scale) {
  if (blockIdx.x == 0 && threadIdx.x == 0) out[0] = acc[0] * scale;
}

// ---------------- host-side orchestration ----------------
struct MP {
  const float *norm_w, *in_proj, *conv_w, *conv_b, *dt_bias, *A_log, *Dp, *gnorm_w, *out_proj, *normf_w;
};
struct Bufs {
  float* xbuf; unsigned short* xnb; float* zx; float* xbcc;
  float* dtv; float* dAv; float* ybuf; unsigned short* ynb;
};

static void launch_convert(const float* s, unsigned short* d, int N, int Kd, int Npad, hipStream_t st) {
  long tot = (long)Npad * Kd;
  convert_pad_kernel<<<(unsigned)((tot + 255) / 256), 256, 0, st>>>(s, d, N, Kd, tot);
}

static void run_stack(const MP& p, unsigned short* ipb, unsigned short* opb,
                      const Bufs& bb, hipStream_t st) {
  for (int l = 0; l < Ll; ++l) {
    rmsnorm_kernel<<<NROWS, 256, 0, st>>>(bb.xbuf, (long)Dd, p.norm_w + (size_t)l * Dd,
                                          bb.xnb, nullptr, Dd);
    {
      int waves = (NROWS / 32) * (NPAD_IN / 64);
      gemm_bf16_wmma<0><<<(waves + 7) / 8, 256, 0, st>>>(
          bb.xnb, ipb + (size_t)l * NPAD_IN * Dd, bb.zx, nullptr,
          NROWS, PROJ, Dd, NPAD_IN, PROJ, 0, 0);
    }
    {
      long tot = (long)Bb * Tt * CONVC;
      conv_silu_kernel<<<(unsigned)((tot + 255) / 256), 256, 0, st>>>(
          bb.zx + INTER, p.conv_w + (size_t)l * CONVC * KC,
          p.conv_b + (size_t)l * CONVC, bb.xbcc, Bb, Tt);
    }
    {
      long tot = (long)NROWS * Hh;
      dtda_kernel<<<(unsigned)((tot + 255) / 256), 256, 0, st>>>(
          bb.zx + INTER + CONVC, p.dt_bias + (size_t)l * Hh, p.A_log + (size_t)l * Hh,
          bb.dtv, bb.dAv, NROWS);
    }
    scan_kernel<<<Bb * Hh, Pp, 0, st>>>(bb.xbcc, bb.dtv, bb.dAv, p.Dp + (size_t)l * Hh,
                                        bb.ybuf, Bb, Tt);
    gated_rmsnorm_kernel<<<NROWS, 256, 0, st>>>(bb.ybuf, bb.zx, PROJ,
                                                p.gnorm_w + (size_t)l * INTER, bb.ynb);
    {
      int waves = (NROWS / 32) * (Dd / 64);
      gemm_bf16_wmma<1><<<(waves + 7) / 8, 256, 0, st>>>(
          bb.ynb, opb + (size_t)l * Dd * INTER, bb.xbuf, nullptr,
          NROWS, Dd, INTER, Dd, Dd, 0, 0);
    }
  }
}

extern "C" void kernel_launch(void* const* d_in, const int* in_sizes, int n_in,
                              void* d_out, int out_size, void* d_ws, size_t ws_size,
                              hipStream_t stream) {
  (void)in_sizes; (void)n_in; (void)out_size; (void)ws_size;

  const float* wte    = (const float*)d_in[0];
  MP enc = { (const float*)d_in[1], (const float*)d_in[2], (const float*)d_in[3],
             (const float*)d_in[4], (const float*)d_in[5], (const float*)d_in[6],
             (const float*)d_in[7], (const float*)d_in[8], (const float*)d_in[9],
             (const float*)d_in[10] };
  MP dec = { (const float*)d_in[11], (const float*)d_in[12], (const float*)d_in[13],
             (const float*)d_in[14], (const float*)d_in[15], (const float*)d_in[16],
             (const float*)d_in[17], (const float*)d_in[18], (const float*)d_in[19],
             (const float*)d_in[20] };
  const float* proj_w    = (const float*)d_in[21];
  const float* proj_b    = (const float*)d_in[22];
  const float* lm_head_w = (const float*)d_in[23];
  const int*   ids       = (const int*)d_in[24];
  const int*   labels    = (const int*)d_in[25];
  float* out = (float*)d_out;

  // -------- carve workspace (fixed, deterministic order) --------
  char* wp = (char*)d_ws;
  auto carve = [&](size_t bytes) -> void* {
    void* p = (void*)wp;
    wp += (bytes + 255) & ~(size_t)255;
    return p;
  };
  float*          xbuf  = (float*)carve((size_t)NROWS * Dd * 4);
  unsigned short* xnb   = (unsigned short*)carve((size_t)NROWS * Dd * 2);
  float*          zx    = (float*)carve((size_t)NROWS * PROJ * 4);
  float*          xbcc  = (float*)carve((size_t)NROWS * CONVC * 4);
  float*          dtv   = (float*)carve((size_t)NROWS * Hh * 4);
  float*          dAv   = (float*)carve((size_t)NROWS * Hh * 4);
  float*          ybuf  = (float*)carve((size_t)NROWS * INTER * 4);
  unsigned short* ynb   = (unsigned short*)carve((size_t)NROWS * INTER * 2);
  float*          ebuf  = (float*)carve((size_t)Bb * Dd * 4);
  unsigned short* winb  = (unsigned short*)carve((size_t)NROWS * D2 * 2);
  unsigned short* ip_e  = (unsigned short*)carve((size_t)Ll * NPAD_IN * Dd * 2);
  unsigned short* op_e  = (unsigned short*)carve((size_t)Ll * Dd * INTER * 2);
  unsigned short* ip_d  = (unsigned short*)carve((size_t)Ll * NPAD_IN * Dd * 2);
  unsigned short* op_d  = (unsigned short*)carve((size_t)Ll * Dd * INTER * 2);
  unsigned short* prjb  = (unsigned short*)carve((size_t)Dd * D2 * 2);
  unsigned short* lmb   = (unsigned short*)carve((size_t)Vv * Dd * 2);
  float*          lacc  = (float*)carve(256);

  // -------- weights -> bf16 (padded rows where needed) --------
  for (int l = 0; l < Ll; ++l) {
    launch_convert(enc.in_proj  + (size_t)l * PROJ * Dd,  ip_e + (size_t)l * NPAD_IN * Dd, PROJ, Dd, NPAD_IN, stream);
    launch_convert(enc.out_proj + (size_t)l * Dd * INTER, op_e + (size_t)l * Dd * INTER,   Dd, INTER, Dd, stream);
    launch_convert(dec.in_proj  + (size_t)l * PROJ * Dd,  ip_d + (size_t)l * NPAD_IN * Dd, PROJ, Dd, NPAD_IN, stream);
    launch_convert(dec.out_proj + (size_t)l * Dd * INTER, op_d + (size_t)l * Dd * INTER,   Dd, INTER, Dd, stream);
  }
  launch_convert(proj_w,    prjb, Dd, D2, Dd, stream);
  launch_convert(lm_head_w, lmb,  Vv, Dd, Vv, stream);

  Bufs bb = { xbuf, xnb, zx, xbcc, dtv, dAv, ybuf, ynb };

  // -------- encoder --------
  {
    long tot = (long)NROWS * Dd;
    embed_kernel<<<(unsigned)((tot + 255) / 256), 256, 0, stream>>>(wte, ids, xbuf, NROWS, Dd);
  }
  run_stack(enc, ip_e, op_e, bb, stream);

  // final RMS of encoder, last token only -> e (B,D)
  rmsnorm_kernel<<<Bb, 256, 0, stream>>>(xbuf + (size_t)(Tt - 1) * Dd, (long)Tt * Dd,
                                         enc.normf_w, nullptr, ebuf, Dd);
  // circular windows -> bf16, then projection (+bias) -> decoder input in xbuf
  {
    long tot = (long)NROWS * D2;
    window_kernel<<<(unsigned)((tot + 255) / 256), 256, 0, stream>>>(ebuf, winb);
  }
  {
    int waves = (NROWS / 32) * (Dd / 64);
    gemm_bf16_wmma<2><<<(waves + 7) / 8, 256, 0, stream>>>(
        winb, prjb, xbuf, proj_b, NROWS, Dd, D2, Dd, Dd, 0, 0);
  }

  // -------- decoder --------
  run_stack(dec, ip_d, op_d, bb, stream);
  rmsnorm_kernel<<<NROWS, 256, 0, stream>>>(xbuf, (long)Dd, dec.normf_w, xnb, nullptr, Dd);

  // -------- lm_head -> transposed logits in d_out[1..], then loss --------
  {
    int waves = (NROWS / 32) * (Vv / 64);
    gemm_bf16_wmma<3><<<(waves + 7) / 8, 256, 0, stream>>>(
        xnb, lmb, out + 1, nullptr, NROWS, Vv, Dd, Vv, 0, Vv, Tt);
  }
  zero_kernel<<<1, 32, 0, stream>>>(lacc);
  loss_kernel<<<NROWS, 256, 0, stream>>>(out + 1, labels, lacc, Vv, Tt);
  finalize_loss<<<1, 32, 0, stream>>>(lacc, out, 1.f / (float)(Bb * Tt));
}
